// GroupedQueryAttention_21784074125379
// MI455X (gfx1250) — matrix operand order repaired and checked by reading, not compiled, not run on hardware
//
#include <hip/hip_runtime.h>
#include <stdint.h>

#define B_  2
#define S_  1024
#define D_  4096
#define H_  32
#define KV_ 8
#define HD_ 128

typedef __attribute__((ext_vector_type(16))) __bf16       v16bf;
typedef __attribute__((ext_vector_type(8)))  float        v8f;
typedef __attribute__((ext_vector_type(4)))  unsigned int u32x4;

#if __has_builtin(__builtin_amdgcn_tensor_load_to_lds)
#define HAVE_TDM 1
typedef __attribute__((ext_vector_type(4))) unsigned int tdm_v4u;
typedef __attribute__((ext_vector_type(8))) int          tdm_v8i;
typedef __attribute__((ext_vector_type(4))) int          tdm_v4i;
#endif

struct Frag32 { u32x4 lo, hi; };

__device__ __forceinline__ unsigned short f2bf(float f) {
  unsigned u = __builtin_bit_cast(unsigned, f);
  u += 0x7FFFu + ((u >> 16) & 1u);          // round-to-nearest-even
  return (unsigned short)(u >> 16);
}
__device__ __forceinline__ float bf2f(unsigned short h) {
  unsigned u = ((unsigned)h) << 16;
  return __builtin_bit_cast(float, u);
}
// truncating bf16 pack of two floats (cheap: 2 VALU) -- used for int8*scale weights
__device__ __forceinline__ unsigned pack2_trunc(float lo, float hi) {
  unsigned ul = __builtin_bit_cast(unsigned, lo);
  unsigned uh = __builtin_bit_cast(unsigned, hi);
  return (ul >> 16) | (uh & 0xffff0000u);
}
__device__ __forceinline__ v16bf make_frag(u32x4 lo, u32x4 hi) {
  Frag32 t; t.lo = lo; t.hi = hi;
  return __builtin_bit_cast(v16bf, t);
}
__device__ __forceinline__ v8f v8zero() {
  v8f z = {0.f, 0.f, 0.f, 0.f, 0.f, 0.f, 0.f, 0.f};
  return z;
}

// A fragment: 16x32 bf16 per CDNA5 ISA 7.12.2 (K half-wave interleave)
__device__ __forceinline__ v16bf frag_a(const unsigned short* t, int ld) {
  int lane = threadIdx.x & 31;
  const unsigned short* p = t + (lane & 15) * ld + (lane >> 4) * 8;
  return make_frag(*(const u32x4*)p, *(const u32x4*)(p + 16));
}
// B fragment: 32x16 bf16, lane n holds column, half-wave holds K=half*8+(0..7), 16+half*8+(0..7)
__device__ __forceinline__ v16bf frag_b(const unsigned short* t, int ld) {
  int lane = threadIdx.x & 31;
  const unsigned short* p = t + (lane & 15) * ld + (lane >> 4) * 8;
  return make_frag(*(const u32x4*)p, *(const u32x4*)(p + 16));
}
__device__ __forceinline__ v8f wmma_bf16(v16bf a, v16bf b, v8f c) {
  return __builtin_amdgcn_wmma_f32_16x16x32_bf16(false, a, false, b, (short)0, c,
                                                 false, false);
}

// ---------------------------------------------------------------- x -> bf16
__global__ __launch_bounds__(256) void cvt_bf16(const float* __restrict__ x,
                                                unsigned short* __restrict__ y, int n) {
  int i = blockIdx.x * blockDim.x + threadIdx.x;
  if (i < n) y[i] = f2bf(x[i]);
}

// ------------------------------------------------------ tile staging (GEMM)
__device__ __forceinline__ void stage_tiles(unsigned short* sA, unsigned short* sB,
                                            int r, int c, u32x4 ra0, u32x4 ra1,
                                            u32x4 rw, float scale) {
  u32x4* da = (u32x4*)(sA + r * 32 + c);
  da[0] = ra0;
  da[1] = ra1;
  const int8_t* pb = (const int8_t*)&rw;
  alignas(16) unsigned packed[8];
#pragma unroll
  for (int i = 0; i < 8; ++i) {
    float lo = (float)pb[2 * i] * scale;
    float hi = (float)pb[2 * i + 1] * scale;
    packed[i] = pack2_trunc(lo, hi);
  }
  u32x4* db = (u32x4*)(sB + r * 32 + c);
  db[0] = ((const u32x4*)packed)[0];
  db[1] = ((const u32x4*)packed)[1];
}

// --------------------------------------------- GEMM: bf16 activ x int8 weights
// Y[M,N] = A[M,K](bf16) * W[N,K](int8)*scale (+bias). 128x128x32 tile,
// double-buffered LDS, 8 waves tiled 4(M)x2(N): 2 A-frags x 4 B-frags, 8 WMMA.
template <bool F32OUT, bool HASBIAS>
__global__ __launch_bounds__(256) void gemm_q8(
    const unsigned short* __restrict__ A, const int8_t* __restrict__ W,
    const float* __restrict__ bias, const float* __restrict__ scale_p,
    unsigned short* __restrict__ outb, float* __restrict__ outf,
    int M, int N, int K) {
  __shared__ alignas(16) unsigned short sA[2][128 * 32];
  __shared__ alignas(16) unsigned short sB[2][128 * 32];
  const float scale = *scale_p;
  const int tid = threadIdx.x, wave = tid >> 5, lane = tid & 31;
  const int wm = wave & 3, wn = wave >> 2;
  const int m0 = blockIdx.x * 128, n0 = blockIdx.y * 128;
  const int r = tid >> 1, c = (tid & 1) * 16;

  const unsigned short* aRow = A + (size_t)(m0 + r) * K + c;
  const int8_t*         wRow = W + (size_t)(n0 + r) * K + c;

  v8f acc[2][4];
#pragma unroll
  for (int i = 0; i < 2; ++i)
#pragma unroll
    for (int j = 0; j < 4; ++j) acc[i][j] = v8zero();

  // preload k-tile 0
  u32x4 ra0 = ((const u32x4*)aRow)[0];
  u32x4 ra1 = ((const u32x4*)aRow)[1];
  u32x4 rw  = *(const u32x4*)wRow;
  stage_tiles(sA[0], sB[0], r, c, ra0, ra1, rw, scale);
  __syncthreads();

  int cur = 0;
  for (int kb = 0; kb < K; kb += 32) {
    const bool hasNext = (kb + 32) < K;
    if (hasNext) {   // issue next tile's global loads before compute
      ra0 = ((const u32x4*)(aRow + kb + 32))[0];
      ra1 = ((const u32x4*)(aRow + kb + 32))[1];
      rw  = *(const u32x4*)(wRow + kb + 32);
    }
    // load all fragments first -> one dscnt wait, then 8 back-to-back WMMAs
    v16bf af[2], bf[4];
#pragma unroll
    for (int i = 0; i < 2; ++i)
      af[i] = frag_a(sA[cur] + (wm * 32 + i * 16) * 32, 32);
#pragma unroll
    for (int j = 0; j < 4; ++j)
      bf[j] = frag_b(sB[cur] + (wn * 64 + j * 16) * 32, 32);
#pragma unroll
    for (int i = 0; i < 2; ++i)
#pragma unroll
      for (int j = 0; j < 4; ++j) acc[i][j] = wmma_bf16(af[i], bf[j], acc[i][j]);

    if (hasNext) {
      stage_tiles(sA[cur ^ 1], sB[cur ^ 1], r, c, ra0, ra1, rw, scale);
      __syncthreads();
      cur ^= 1;
    }
  }

  // epilogue: C layout -> row = q + 8*half, col = lane&15
  const int hf = lane >> 4, ln = lane & 15;
#pragma unroll
  for (int i = 0; i < 2; ++i) {
    int rbase = m0 + wm * 32 + i * 16 + hf * 8;
#pragma unroll
    for (int j = 0; j < 4; ++j) {
      int col = n0 + wn * 64 + j * 16 + ln;
      float bv = HASBIAS ? bias[col] : 0.f;
#pragma unroll
      for (int q = 0; q < 8; ++q) {
        float v = acc[i][j][q] + bv;
        if (F32OUT) outf[(size_t)(rbase + q) * N + col] = v;
        else        outb[(size_t)(rbase + q) * N + col] = f2bf(v);
      }
    }
  }
}

// ---------------------------------------------------------------- RoPE (Q,K)
__global__ __launch_bounds__(256) void rope_k(unsigned short* __restrict__ q,
                                              unsigned short* __restrict__ k,
                                              const float* __restrict__ cosb,
                                              const float* __restrict__ sinb, int n) {
  int i = blockIdx.x * blockDim.x + threadIdx.x;
  if (i >= n) return;
  const int nq = B_ * S_ * H_ * 64;
  unsigned short* p;
  int s, hd;
  if (i < nq) {
    hd = i & 63; int h = (i >> 6) & 31; s = (i >> 11) & 1023; int b = i >> 21;
    p = q + ((size_t)(b * S_ + s) * H_ + h) * HD_ + hd;
  } else {
    int j = i - nq;
    hd = j & 63; int h = (j >> 6) & 7; s = (j >> 9) & 1023; int b = j >> 19;
    p = k + ((size_t)(b * S_ + s) * KV_ + h) * HD_ + hd;
  }
  float c0 = cosb[s * HD_ + hd],      s0 = sinb[s * HD_ + hd];
  float c1 = cosb[s * HD_ + hd + 64], s1 = sinb[s * HD_ + hd + 64];
  float x1 = bf2f(p[0]), x2 = bf2f(p[64]);
  p[0]  = f2bf(x1 * c0 - x2 * s0);
  p[64] = f2bf(x2 * c1 + x1 * s1);
}

// ------------------------------------------------- flash attention (causal GQA)
// block = 4 waves, one (b,h,64-row q tile). wave owns 16 q rows.
__global__ __launch_bounds__(128) void attn_fa(
    const unsigned short* __restrict__ Q,   // [B,S,H,HD] bf16
    const unsigned short* __restrict__ Kc,  // [B,S,KV,HD] bf16
    const unsigned short* __restrict__ Vc,  // [B,S,KV,HD] bf16
    unsigned short* __restrict__ O) {       // [B,S,H,HD] bf16
  __shared__ alignas(16) unsigned short sK[32 * 128];    // kv-rows x hd
  __shared__ alignas(16) unsigned short sVt[128 * 32];   // hd x kv (pair-packed)
  __shared__ alignas(16) unsigned short sP[4][16 * 32];  // per-wave P tile
  const int bid = blockIdx.x;
  const int qt = bid & 15;                 // S/64 q tiles
  const int h  = (bid >> 4) & 31;
  const int b  = bid >> 9;
  const int kvh = h >> 2;                  // GROUP = 4
  const int q0 = qt * 64;
  const int tid = threadIdx.x, wave = tid >> 5, lane = tid & 31;
  const int m = lane & 15, hf = lane >> 4;
  const int mrow0 = q0 + wave * 16;

  // Q fragments: 16 rows x 128 k -> 4 A-frags in VGPRs
  const unsigned short* qp = Q + ((size_t)(b * S_ + mrow0 + m) * H_ + h) * HD_;
  v16bf qf[4];
#pragma unroll
  for (int kk = 0; kk < 4; ++kk) {
    const unsigned short* p = qp + kk * 32 + hf * 8;
    qf[kk] = make_frag(*(const u32x4*)p, *(const u32x4*)(p + 16));
  }

  float mr[8], lr[8];
  v8f o[8];
#pragma unroll
  for (int j = 0; j < 8; ++j) { mr[j] = -1e30f; lr[j] = 0.f; }
#pragma unroll
  for (int i = 0; i < 8; ++i) o[i] = v8zero();

  const float SC = 0.08838834764831845f;   // 1/sqrt(128)

  for (int k0 = 0; k0 < q0 + 64; k0 += 32) {   // causal: skip upper tiles
    // ---- stage K tile (32x128) via Tensor Data Mover (wave 0 issues DMA) ----
#ifdef HAVE_TDM
    if (wave == 0) {
      unsigned lds_off = (unsigned)(uintptr_t)&sK[0];   // low 32b = LDS offset
      unsigned long long ga =
          (unsigned long long)(uintptr_t)(Kc + ((size_t)(b * S_ + k0) * KV_ + kvh) * HD_);
      // D# group0: count=1, lds_addr, global_addr[56:0], type=2
      tdm_v4u g0 = { 1u, lds_off, (unsigned)(ga & 0xffffffffu),
                     (unsigned)(((ga >> 32) & 0x01ffffffu) | 0x80000000u) };
      // D# group1: data_size=2B, tensor_dim0=128, tensor_dim1=rows remaining,
      //            tile 128x32, tensor_dim0_stride=KV*HD
      unsigned td1 = (unsigned)(S_ * KV_ - k0);   // rows remaining (OOB guard)
      tdm_v8i g1 = { (int)0x00010000,
                     (int)(128u << 16),
                     (int)((td1 & 0xffffu) << 16),
                     (int)((td1 >> 16) | (128u << 16)),
                     (int)32, (int)(KV_ * HD_), 0, 0 };
      tdm_v4i gz4 = { 0, 0, 0, 0 };
      tdm_v8i gz8 = { 0, 0, 0, 0, 0, 0, 0, 0 };
      __builtin_amdgcn_tensor_load_to_lds(g0, g1, gz4, gz4, gz8, 0);
#if __has_builtin(__builtin_amdgcn_s_wait_tensorcnt)
      __builtin_amdgcn_s_wait_tensorcnt(0);
#else
      asm volatile("s_wait_tensorcnt 0x0" ::: "memory");
#endif
    }
#else
    {
      int rr = tid >> 2, cc = (tid & 3) * 32;
      const unsigned short* kp =
          Kc + ((size_t)(b * S_ + k0 + rr) * KV_ + kvh) * HD_ + cc;
      u32x4* dst = (u32x4*)(sK + rr * 128 + cc);
      const u32x4* src = (const u32x4*)kp;
      dst[0] = src[0]; dst[1] = src[1]; dst[2] = src[2]; dst[3] = src[3];
    }
#endif
    // ---- stage V transposed, kv-pairs packed into b32 ----
    {
      int r2 = tid >> 3, c16 = (tid & 7) * 16;   // kv rows 2r2,2r2+1 ; 16 hd cols
      const unsigned short* v0p =
          Vc + ((size_t)(b * S_ + k0 + 2 * r2) * KV_ + kvh) * HD_ + c16;
      const unsigned short* v1p = v0p + KV_ * HD_;
      u32x4 ra = *(const u32x4*)v0p, rb = *(const u32x4*)(v0p + 8);
      u32x4 rc = *(const u32x4*)v1p, rd = *(const u32x4*)(v1p + 8);
      unsigned* sv32 = (unsigned*)sVt;           // [hd][16] kv-pair words
#pragma unroll
      for (int i = 0; i < 4; ++i) {
        unsigned a = ra[i], cc2 = rc[i];
        sv32[(c16 + 2 * i) * 16 + r2]     = (a & 0xffffu) | (cc2 << 16);
        sv32[(c16 + 2 * i + 1) * 16 + r2] = (a >> 16) | (cc2 & 0xffff0000u);
        unsigned b2 = rb[i], d2 = rd[i];
        sv32[(c16 + 8 + 2 * i) * 16 + r2]     = (b2 & 0xffffu) | (d2 << 16);
        sv32[(c16 + 8 + 2 * i + 1) * 16 + r2] = (b2 >> 16) | (d2 & 0xffff0000u);
      }
    }
    __syncthreads();

    // ---- scores: load all 8 K-frags, then 8 WMMAs ----
    v16bf kf[8];
#pragma unroll
    for (int t = 0; t < 2; ++t)
#pragma unroll
      for (int kk = 0; kk < 4; ++kk)
        kf[t * 4 + kk] = frag_b(sK + t * 16 * 128 + kk * 32, 128);
    v8f sc[2];
    sc[0] = v8zero(); sc[1] = v8zero();
#pragma unroll
    for (int t = 0; t < 2; ++t)
#pragma unroll
      for (int kk = 0; kk < 4; ++kk)
        sc[t] = wmma_bf16(qf[kk], kf[t * 4 + kk], sc[t]);

    // ---- online softmax (row reductions across the 16-lane N group) ----
#pragma unroll
    for (int j = 0; j < 8; ++j) {
      int row = mrow0 + j + 8 * hf;
      float v0 = sc[0][j] * SC;
      float v1 = sc[1][j] * SC;
      if (k0 + m > row)      v0 = -1e30f;
      if (k0 + 16 + m > row) v1 = -1e30f;
      float mx = fmaxf(v0, v1);
#pragma unroll
      for (int d = 1; d < 16; d <<= 1) mx = fmaxf(mx, __shfl_xor(mx, d, 32));
      float mn = fmaxf(mr[j], mx);
      float alpha = __expf(mr[j] - mn);
      mr[j] = mn;
      float p0 = __expf(v0 - mn);
      float p1 = __expf(v1 - mn);
      float rs = p0 + p1;
#pragma unroll
      for (int d = 1; d < 16; d <<= 1) rs += __shfl_xor(rs, d, 32);
      lr[j] = lr[j] * alpha + rs;
#pragma unroll
      for (int nt = 0; nt < 8; ++nt) o[nt][j] *= alpha;
      int prow = (j + 8 * hf) * 32;
      sP[wave][prow + m]      = f2bf(p0);
      sP[wave][prow + 16 + m] = f2bf(p1);
    }

    // ---- O += P(16x32) x V(32x128): 8 WMMAs ----
    v16bf pf = frag_a(sP[wave], 32);
    v16bf vf[8];
#pragma unroll
    for (int nt = 0; nt < 8; ++nt) vf[nt] = frag_b(sVt + nt * 16 * 32, 32);
#pragma unroll
    for (int nt = 0; nt < 8; ++nt) o[nt] = wmma_bf16(pf, vf[nt], o[nt]);
    __syncthreads();
  }

  // normalize and store [b,s,h,hd]
#pragma unroll
  for (int nt = 0; nt < 8; ++nt)
#pragma unroll
    for (int j = 0; j < 8; ++j) {
      int row = mrow0 + j + 8 * hf;
      float v = o[nt][j] / lr[j];
      O[((size_t)(b * S_ + row) * H_ + h) * HD_ + nt * 16 + m] = f2bf(v);
    }
}

// ---------------------------------------------------------------------- launch
extern "C" void kernel_launch(void* const* d_in, const int* in_sizes, int n_in,
                              void* d_out, int out_size, void* d_ws, size_t ws_size,
                              hipStream_t stream) {
  const float*  x    = (const float*)d_in[0];
  const float*  cosb = (const float*)d_in[2];
  const float*  sinb = (const float*)d_in[3];
  const int8_t* wq   = (const int8_t*)d_in[4];
  const float*  bq   = (const float*)d_in[5];
  const float*  sq   = (const float*)d_in[6];
  const int8_t* wk   = (const int8_t*)d_in[7];
  const float*  bk   = (const float*)d_in[8];
  const float*  sk   = (const float*)d_in[9];
  const int8_t* wv   = (const int8_t*)d_in[10];
  const float*  bv   = (const float*)d_in[11];
  const float*  sv   = (const float*)d_in[12];
  const int8_t* wo   = (const int8_t*)d_in[13];
  const float*  so   = (const float*)d_in[14];
  float* out = (float*)d_out;

  unsigned short* xb = (unsigned short*)d_ws;                 // B*S*D
  unsigned short* qb = xb + (size_t)B_ * S_ * D_;             // B*S*H*HD
  unsigned short* kb = qb + (size_t)B_ * S_ * H_ * HD_;       // B*S*KV*HD
  unsigned short* vb = kb + (size_t)B_ * S_ * KV_ * HD_;      // B*S*KV*HD
  unsigned short* ao = vb + (size_t)B_ * S_ * KV_ * HD_;      // B*S*H*HD

  const int M = B_ * S_;
  {
    int n = B_ * S_ * D_;
    cvt_bf16<<<(n + 255) / 256, 256, 0, stream>>>(x, xb, n);
  }
  gemm_q8<false, true><<<dim3(M / 128, (H_ * HD_) / 128), 256, 0, stream>>>(
      xb, wq, bq, sq, qb, nullptr, M, H_ * HD_, D_);
  gemm_q8<false, true><<<dim3(M / 128, (KV_ * HD_) / 128), 256, 0, stream>>>(
      xb, wk, bk, sk, kb, nullptr, M, KV_ * HD_, D_);
  gemm_q8<false, true><<<dim3(M / 128, (KV_ * HD_) / 128), 256, 0, stream>>>(
      xb, wv, bv, sv, vb, nullptr, M, KV_ * HD_, D_);
  {
    int n = B_ * S_ * H_ * 64 + B_ * S_ * KV_ * 64;
    rope_k<<<(n + 255) / 256, 256, 0, stream>>>(qb, kb, cosb, sinb, n);
  }
  attn_fa<<<B_ * H_ * (S_ / 64), 128, 0, stream>>>(qb, kb, vb, ao);
  gemm_q8<true, false><<<dim3(M / 128, D_ / 128), 256, 0, stream>>>(
      ao, wo, nullptr, so, nullptr, out, M, D_, H_ * HD_);
}
